// GaussianQuant_58952721105342
// MI455X (gfx1250) — matrix operand order repaired and checked by reading, not compiled, not run on hardware
//
#include <hip/hip_runtime.h>

// ---------------------------------------------------------------------------
// GaussianQuant forward for MI455X (gfx1250, wave32, WMMA + async-to-LDS).
//
// score[n,k] = (mu*invvar)_n . c_k + (-0.5*invvar)_n . c2_k + bias[k]
// -> 16384 x 16384 GEMM, reduction depth 64, fused streaming argmax.
// v_wmma_f32_16x16x32_bf16: 2 matrix instrs per 16x16 tile; bias folded into
// the C operand; codebook double-buffered through LDS with
// global_load_async_to_lds_b128 (ASYNCcnt) so staging overlaps compute.
// ---------------------------------------------------------------------------

typedef __attribute__((ext_vector_type(16))) __bf16 v16bf;
typedef __attribute__((ext_vector_type(8)))  float  v8f;

union FragBF { v16bf v; uint4 q[2]; };

#define HALF_LOG_2PI 0.9189385332046727f
#define ZHAT_ELEMS 524288 // 4*512*256

__device__ __forceinline__ unsigned short f2bf(float f) {
    unsigned int u = __float_as_uint(f);
    unsigned int r = u + 0x7FFFu + ((u >> 16) & 1u);   // round-to-nearest-even
    return (unsigned short)(r >> 16);
}

__device__ __forceinline__ void async_ld16(unsigned lds_off, unsigned long long gaddr) {
    // CDNA5 async global->LDS copy, tracked by ASYNCcnt (no VGPR round-trip)
    asm volatile("global_load_async_to_lds_b128 %0, %1, off"
                 :: "v"(lds_off), "v"(gaddr) : "memory");
}
__device__ __forceinline__ void wait_async0() {
    asm volatile("s_wait_asynccnt 0x0" ::: "memory");
}

// ---- prep 1: rows -> X fragments (bf16) + KL reduction -------------------
// X layout: row n (= t*8+c), 64 bf16: [0..31]=mu*invvar, [32..63]=-0.5*invvar
__global__ __launch_bounds__(256) void gq_prep_rows(
    const float* __restrict__ z, unsigned short* __restrict__ Xbf,
    float* __restrict__ klsum)
{
    int idx = blockIdx.x * 256 + threadIdx.x;     // 524288 threads
    int t = idx >> 8;                             // token 0..2047
    int j = idx & 255;                            // 0..255 = d*8+c
    int d = j >> 3;
    int c = j & 7;
    int n = t * 8 + c;

    float mu = z[t * 512 + j];
    float lv = z[t * 512 + 256 + j];
    lv = fminf(fmaxf(lv, -30.0f), 20.0f);
    float var    = __expf(lv);
    float invvar = __expf(-lv);

    Xbf[n * 64 + d]      = f2bf(mu * invvar);
    Xbf[n * 64 + 32 + d] = f2bf(-0.5f * invvar);

    // kl2 term: 1.4426*0.5*(mu^2 + var - 1 - logvar); kl_loss = mean of row sums
    float term = 0.7213f * (mu * mu + var - 1.0f - lv);
    #pragma unroll
    for (int off = 16; off > 0; off >>= 1)
        term += __shfl_down(term, off);
    if ((threadIdx.x & 31) == 0)
        atomicAdd(klsum, term);
}

// ---- prep 2: codebook -> Y fragments (bf16) + fp32 bias -------------------
// Y layout: row k, 64 bf16: [0..31]=c, [32..63]=c^2.  bias[k] = -prior_lp[k].
__global__ __launch_bounds__(256) void gq_prep_cb(
    const float* __restrict__ cb, unsigned short* __restrict__ Ybf,
    float* __restrict__ bias)
{
    int idx = blockIdx.x * 256 + threadIdx.x;     // 524288 threads
    int k = idx >> 5;
    int d = idx & 31;                             // == lane (wave32)
    float c = cb[k * 32 + d];
    Ybf[k * 64 + d]      = f2bf(c);
    Ybf[k * 64 + 32 + d] = f2bf(c * c);
    float red = 0.5f * c * c;
    #pragma unroll
    for (int off = 16; off > 0; off >>= 1)
        red += __shfl_down(red, off);
    if (d == 0)
        bias[k] = red + 32.0f * HALF_LOG_2PI;
}

__global__ void gq_finalize_kl(const float* __restrict__ klsum,
                               float* __restrict__ out)
{
    out[ZHAT_ELEMS] = klsum[0] * (1.0f / 16384.0f);
}

// ---- main: fused GEMM + argmax + gather --------------------------------
// Block = 128 threads = 4 waves; each wave owns one 16-row n-tile.
// Codebook double-buffered 8 k-tiles (16 KB) at a time via async-to-LDS.
__global__ __launch_bounds__(128) void gq_search(
    const unsigned short* __restrict__ Xbf,
    const unsigned short* __restrict__ Ybf,
    const float* __restrict__ bias,
    const float* __restrict__ codebook,
    float* __restrict__ out)
{
    __shared__ uint4 lbuf[2][1024];               // 2 x 16 KB double buffer
    const int tid  = threadIdx.x;
    const int lane = tid & 31;
    const int wave = tid >> 5;
    const int half = lane >> 4;                   // lane-half per ISA layouts
    const int m16  = lane & 15;
    const int ntile = blockIdx.x * 4 + wave;      // 256 blocks * 4 waves = 1024

    // B operand: X row (N = token-row), loaded once per wave.
    // 16-bit B layout: lanes0-15 K=0-7 & 16-23; lanes16-31 K=8-15 & 24-31.
    const uint4* Xq = (const uint4*)Xbf;          // 8 uint4 per 128B row
    const int xrow = ntile * 16 + m16;
    FragBF fb0, fb1;
    fb0.q[0] = Xq[xrow * 8 + half];
    fb0.q[1] = Xq[xrow * 8 + 2 + half];
    fb1.q[0] = Xq[xrow * 8 + 4 + half];
    fb1.q[1] = Xq[xrow * 8 + 6 + half];

    const uint4*  Yq = (const uint4*)Ybf;
    const float4* bq = (const float4*)bias;

    // async stage of one 16 KB chunk (128 codebook rows), XOR-swizzled banks
    auto stage = [&](int chunk, int buf) {
        const uint4* src = Yq + chunk * 1024;
        #pragma unroll
        for (int i = 0; i < 8; ++i) {
            int j  = tid + i * 128;
            int r  = j >> 3;
            int cw = j & 7;
            unsigned dst = (unsigned)(unsigned long long)
                               (&lbuf[buf][(r << 3) | (cw ^ (r & 7))]);
            async_ld16(dst, (unsigned long long)(src + j));
        }
    };
    // A operand: codebook rows (M = k within tile).
    // 16-bit A layout: lanes0-15 K=0-7 & 16-23; lanes16-31 K=8-15 & 24-31.
    auto loadA = [&](int buf, int tt, FragBF& fa0, FragBF& fa1) {
        const int grow = tt * 16 + m16;
        const int base = grow << 3;
        const int sw   = grow & 7;
        fa0.q[0] = lbuf[buf][base | ((0 + half) ^ sw)];
        fa0.q[1] = lbuf[buf][base | ((2 + half) ^ sw)];
        fa1.q[0] = lbuf[buf][base | ((4 + half) ^ sw)];
        fa1.q[1] = lbuf[buf][base | ((6 + half) ^ sw)];
    };
    // bias folded into WMMA C operand: acc slot v holds k = kbase + half*8 + v
    auto biasInit = [&](int chunk, int tt) -> v8f {
        const int kbase = (chunk * 8 + tt) * 16;
        float4 b0 = bq[(kbase >> 2) + half * 2];
        float4 b1 = bq[(kbase >> 2) + half * 2 + 1];
        v8f a;
        a[0] = b0.x; a[1] = b0.y; a[2] = b0.z; a[3] = b0.w;
        a[4] = b1.x; a[5] = b1.y; a[6] = b1.z; a[7] = b1.w;
        return a;
    };

    float bestS[8];
    int   bestK[8];
    #pragma unroll
    for (int v = 0; v < 8; ++v) { bestS[v] = -3.0e38f; bestK[v] = 0; }

    auto update = [&](const v8f& acc, int chunk, int tt) {
        const int k0 = (chunk * 8 + tt) * 16 + half * 8;
        #pragma unroll
        for (int v = 0; v < 8; ++v) {
            float s = acc[v];
            bool better = s > bestS[v];
            bestS[v] = better ? s : bestS[v];
            bestK[v] = better ? (k0 + v) : bestK[v];
        }
    };

    stage(0, 0);
    wait_async0();
    __syncthreads();

    for (int chunk = 0; chunk < 128; ++chunk) {
        const int buf = chunk & 1;
        if (chunk + 1 < 128)
            stage(chunk + 1, buf ^ 1);            // overlaps with compute below

        // 8 k-tiles, software-pipelined one tile deep so argmax VALU work
        // co-executes with the next tile's WMMAs (hides WMMA->VALU hazard)
        FragBF fa0, fa1;
        loadA(buf, 0, fa0, fa1);
        v8f accP = biasInit(chunk, 0);
        accP = __builtin_amdgcn_wmma_f32_16x16x32_bf16(
                   false, fa0.v, false, fb0.v, (short)0, accP, false, false);
        accP = __builtin_amdgcn_wmma_f32_16x16x32_bf16(
                   false, fa1.v, false, fb1.v, (short)0, accP, false, false);
        #pragma unroll
        for (int tt = 1; tt < 8; ++tt) {
            FragBF ga0, ga1;
            loadA(buf, tt, ga0, ga1);
            v8f accN = biasInit(chunk, tt);
            accN = __builtin_amdgcn_wmma_f32_16x16x32_bf16(
                       false, ga0.v, false, fb0.v, (short)0, accN, false, false);
            accN = __builtin_amdgcn_wmma_f32_16x16x32_bf16(
                       false, ga1.v, false, fb1.v, (short)0, accN, false, false);
            update(accP, chunk, tt - 1);
            accP = accN;
        }
        update(accP, chunk, 7);

        wait_async0();                            // next chunk landed in LDS
        __syncthreads();
    }

    // per-lane slot reduction (prefer smallest k on ties, like jnp.argmax)
    float bS = bestS[0];
    int   bK = bestK[0];
    #pragma unroll
    for (int v = 1; v < 8; ++v) {
        bool better = (bestS[v] > bS) || (bestS[v] == bS && bestK[v] < bK);
        bS = better ? bestS[v] : bS;
        bK = better ? bestK[v] : bK;
    }
    // merge lane L with lane L^16 (same token row)
    float oS = __shfl_xor(bS, 16);
    int   oK = __shfl_xor(bK, 16);
    {
        bool better = (oS > bS) || (oS == bS && oK < bK);
        bS = better ? oS : bS;
        bK = better ? oK : bK;
    }

    if (lane < 16) {
        const int g = ntile * 16 + lane;          // global row n = t*8 + c
        const int t = g >> 3, c = g & 7;
        const float* crow = codebook + bK * 32;
        #pragma unroll
        for (int d = 0; d < 32; ++d)
            out[t * 256 + d * 8 + c] = crow[d];   // zhat (== zhat_v via STE)
        out[ZHAT_ELEMS + 1 + g] = (float)bK;      // indices
    }
}

// ---------------------------------------------------------------------------
extern "C" void kernel_launch(void* const* d_in, const int* in_sizes, int n_in,
                              void* d_out, int out_size, void* d_ws, size_t ws_size,
                              hipStream_t stream) {
    const float* z     = (const float*)d_in[0];
    const float* noise = (const float*)d_in[1]; (void)noise; // dead in forward (STE)
    const float* cb    = (const float*)d_in[2];
    float* out = (float*)d_out;

    char* ws = (char*)d_ws;
    float*          klsum = (float*)ws;                          // 4 B
    float*          bias  = (float*)(ws + 256);                  // 64 KB
    unsigned short* Xbf   = (unsigned short*)(ws + 256 + 65536);            // 2 MB
    unsigned short* Ybf   = (unsigned short*)(ws + 256 + 65536 + 2097152);  // 2 MB

    hipMemsetAsync(klsum, 0, sizeof(float), stream);
    gq_prep_rows<<<2048, 256, 0, stream>>>(z, Xbf, klsum);
    gq_prep_cb  <<<2048, 256, 0, stream>>>(cb, Ybf, bias);
    gq_finalize_kl<<<1, 1, 0, stream>>>(klsum, out);
    gq_search<<<256, 128, 0, stream>>>(Xbf, Ybf, bias, cb, out);
}
